// Attention_70111046140627
// MI455X (gfx1250) — compile-verified, matrix operand order checked by reading, not run on hardware
//
#include <hip/hip_runtime.h>

// Problem constants (from reference): B=2, S=2048, D=1024, H=16, DK=64
#define BSZ    2
#define SLEN   2048
#define DMODEL 1024
#define NHEAD  16
#define DKH    64

typedef __attribute__((ext_vector_type(16))) __bf16 v16bf;
typedef __attribute__((ext_vector_type(8)))  float  v8f;
typedef unsigned int __attribute__((ext_vector_type(4))) u32x4;
typedef int  __attribute__((ext_vector_type(8))) i32x8;
typedef int  __attribute__((ext_vector_type(4))) i32x4;

#if defined(__has_builtin)
#if __has_builtin(__builtin_amdgcn_tensor_load_to_lds) && \
    __has_builtin(__builtin_amdgcn_s_wait_tensorcnt)
#define USE_TDM 1
#endif
#endif

union FragA {          // 16x32 bf16 A/B fragment (8 VGPRs)
    v16bf v;
    __bf16 h[16];
    uint4  q[2];
};
union FragC {          // 16x16 f32 C/D fragment (8 VGPRs)
    v8f   v;
    float f[8];
};

// Per documented 16-bit A-matrix layout: lane holds row M = lane%16; element e
// maps to K = (e<8 ? half*8 + e : 16 + half*8 + (e-8)), i.e. two contiguous
// 8-element (16-byte) runs at [half*8, +8) and [16+half*8, +8).

// ---------------------------------------------------------------------------
// Kernel 1: out = x @ W^T + b, written as bf16 [which][B][H][S][DK]
// block = 128 threads (4 waves); block tile = 64 rows x 64 cols; K step 32.
// ---------------------------------------------------------------------------
__global__ __launch_bounds__(128) void qkv_proj_kernel(
    const float* __restrict__ x,
    const float* __restrict__ Wq, const float* __restrict__ bq,
    const float* __restrict__ Wk, const float* __restrict__ bk,
    const float* __restrict__ Wv, const float* __restrict__ bv,
    __bf16* __restrict__ qkv)
{
    const int which = blockIdx.z;
    const float* W    = (which == 0) ? Wq : (which == 1) ? Wk : Wv;
    const float* bias = (which == 0) ? bq : (which == 1) ? bk : bv;
    __bf16* dst = qkv + (size_t)which * BSZ * NHEAD * SLEN * DKH;

    __shared__ __align__(16) __bf16 Xs[64][32];
    __shared__ __align__(16) __bf16 Ws[64][32];

    const int tid  = threadIdx.x;
    const int wave = tid >> 5;
    const int lane = tid & 31;
    const int m16  = lane & 15;
    const int half = lane >> 4;

    const int mBase = blockIdx.x * 64;   // row tile in [B*S]
    const int nBase = blockIdx.y * 64;   // col tile in [D]

    FragC acc[4];
    #pragma unroll
    for (int g = 0; g < 4; ++g)
        #pragma unroll
        for (int i = 0; i < 8; ++i) acc[g].f[i] = 0.0f;

    for (int k0 = 0; k0 < DMODEL; k0 += 32) {
        __syncthreads();
        // Stage X tile (64x32) and W tile (64x32) as bf16 in LDS.
        #pragma unroll
        for (int j = 0; j < 16; j += 4) {
            int flat = tid * 16 + j;             // 0..2047
            int row  = flat >> 5;
            int col  = flat & 31;
            float4 xv = *reinterpret_cast<const float4*>(
                &x[(size_t)(mBase + row) * DMODEL + k0 + col]);
            Xs[row][col + 0] = (__bf16)xv.x;
            Xs[row][col + 1] = (__bf16)xv.y;
            Xs[row][col + 2] = (__bf16)xv.z;
            Xs[row][col + 3] = (__bf16)xv.w;
            float4 wv = *reinterpret_cast<const float4*>(
                &W[(size_t)(nBase + row) * DMODEL + k0 + col]);
            Ws[row][col + 0] = (__bf16)wv.x;
            Ws[row][col + 1] = (__bf16)wv.y;
            Ws[row][col + 2] = (__bf16)wv.z;
            Ws[row][col + 3] = (__bf16)wv.w;
        }
        __syncthreads();

        // A fragment: this wave's 16 rows.
        FragA a;
        {
            int row = wave * 16 + m16;
            a.q[0] = *reinterpret_cast<const uint4*>(&Xs[row][half * 8]);
            a.q[1] = *reinterpret_cast<const uint4*>(&Xs[row][16 + half * 8]);
        }
        #pragma unroll
        for (int g = 0; g < 4; ++g) {
            FragA bf;
            int row = g * 16 + m16;              // W row == B^T row
            bf.q[0] = *reinterpret_cast<const uint4*>(&Ws[row][half * 8]);
            bf.q[1] = *reinterpret_cast<const uint4*>(&Ws[row][16 + half * 8]);
            acc[g].v = __builtin_amdgcn_wmma_f32_16x16x32_bf16(
                false, a.v, false, bf.v, (short)0, acc[g].v, false, false);
        }
    }

    // Epilogue: + bias, store bf16 into [B][H][S][DK].
    #pragma unroll
    for (int g = 0; g < 4; ++g) {
        int n  = nBase + g * 16 + m16;
        float bvv = bias[n];
        int hh = n >> 6;
        int dk = n & 63;
        #pragma unroll
        for (int r = 0; r < 8; ++r) {
            int m = mBase + wave * 16 + r + half * 8;
            int b = m >> 11;                     // / SLEN
            int s = m & (SLEN - 1);
            float val = acc[g].f[r] + bvv;
            dst[(((size_t)b * NHEAD + hh) * SLEN + s) * DKH + dk] = (__bf16)val;
        }
    }
}

// ---------------------------------------------------------------------------
// Kernel 2: causal flash attention. Block = 4 waves = 64 q rows of one (b,h).
// K tile staged by the Tensor Data Mover (one descriptor per iteration issued
// from wave 0, overlapped with the VALU-side V-transpose staging).
// ---------------------------------------------------------------------------
__global__ __launch_bounds__(128) void flash_attn_kernel(
    const __bf16* __restrict__ qkv, float* __restrict__ out)
{
    const int tid  = threadIdx.x;
    const int wave = tid >> 5;
    const int lane = tid & 31;
    const int m16  = lane & 15;
    const int half = lane >> 4;

    const int bh = blockIdx.y;          // 0..B*H-1
    const int b  = bh >> 4;
    const int h  = bh & 15;
    const int qt = blockIdx.x;          // 64-row q tile index
    const int qbase = qt * 64 + wave * 16;

    const size_t planeSz = (size_t)BSZ * NHEAD * SLEN * DKH;
    const size_t headOff = ((size_t)(b * NHEAD + h)) * SLEN * DKH;
    const __bf16* qp = qkv + headOff;
    const __bf16* kp = qkv + planeSz + headOff;
    const __bf16* vp = qkv + 2 * planeSz + headOff;

    __shared__ __align__(16) __bf16 Ksh[32][64];     // [key][dk]
    __shared__ __align__(16) __bf16 VTsh[64][32];    // [dk][key] (transposed)
    __shared__ __align__(16) __bf16 Psh[4][16][32];  // per-wave P scratch

    // Q fragments (dk 0..31 and 32..63), loaded once.
    FragA qa[2];
    {
        const __bf16* qrow = qp + (size_t)(qbase + m16) * DKH;
        #pragma unroll
        for (int t = 0; t < 2; ++t) {
            qa[t].q[0] = *reinterpret_cast<const uint4*>(&qrow[t * 32 + half * 8]);
            qa[t].q[1] = *reinterpret_cast<const uint4*>(&qrow[t * 32 + 16 + half * 8]);
        }
    }

    FragC acc[4];
    #pragma unroll
    for (int g = 0; g < 4; ++g)
        #pragma unroll
        for (int i = 0; i < 8; ++i) acc[g].f[i] = 0.0f;

    float mrow[8], lrow[8];
    #pragma unroll
    for (int r = 0; r < 8; ++r) { mrow[r] = -1e30f; lrow[r] = 0.0f; }

#ifdef USE_TDM
    // Tensor DMA descriptor pieces that are loop-invariant (D# per ISA 8.3-8.5).
    const unsigned int ldsK = (unsigned int)(size_t)&Ksh[0][0]; // flat->LDS: addr[31:0]
    i32x8 dg1;
    dg1[0] = 0x00010000;        // workgroup_mask=0, data_size=1 (2B), no flags
    dg1[1] = (DKH << 16);       // atomic_barrier_addr=0; tensor_dim0[15:0]=64
    dg1[2] = (32  << 16);       // tensor_dim0[31:16]=0; tensor_dim1[15:0]=32
    dg1[3] = (DKH << 16);       // tensor_dim1[31:16]=0; tile_dim0=64
    dg1[4] = 32;                // tile_dim1=32 keys; tile_dim2=0
    dg1[5] = DKH;               // tensor_dim0_stride[31:0]=64
    dg1[6] = 0;                 // stride0[47:32]=0; tensor_dim1_stride[15:0]=0
    dg1[7] = 0;                 // (2D tile: dim1 stride unused)
    const i32x4 dg23 = {0, 0, 0, 0};
    const i32x8 dgx  = {0, 0, 0, 0, 0, 0, 0, 0};
#endif

    const int kEnd = qt * 64 + 64;      // causal upper bound for this block
    for (int kt = 0; kt < kEnd; kt += 32) {
        __syncthreads();
#ifdef USE_TDM
        // K tile [32][64] via Tensor Data Mover, issued by wave 0.
        if (wave == 0) {
            unsigned long long ga =
                (unsigned long long)(size_t)(kp + (size_t)kt * DKH);
            u32x4 dg0;
            dg0[0] = 1u;                                    // count=1, user mode
            dg0[1] = ldsK;                                  // lds_addr
            dg0[2] = (unsigned int)ga;                      // global_addr[31:0]
            dg0[3] = ((unsigned int)(ga >> 32) & 0x01FFFFFFu) | 0x80000000u; // type=2
            __builtin_amdgcn_tensor_load_to_lds(dg0, dg1, dg23, dg23, dgx, 0);
        }
#endif
        // Stage V^T [64][32] into LDS (all threads).
        #pragma unroll
        for (int j = 0; j < 16; j += 8) {
            int flat = tid * 16 + j;
            int key  = flat >> 6;
            int dk   = flat & 63;
#ifndef USE_TDM
            uint4 kv = *reinterpret_cast<const uint4*>(
                &kp[(size_t)(kt + key) * DKH + dk]);
            *reinterpret_cast<uint4*>(&Ksh[key][dk]) = kv;
#endif
            uint4 vv = *reinterpret_cast<const uint4*>(
                &vp[(size_t)(kt + key) * DKH + dk]);
            const __bf16* ve = reinterpret_cast<const __bf16*>(&vv);
            #pragma unroll
            for (int i = 0; i < 8; ++i) VTsh[dk + i][key] = ve[i];
        }
        if (kt + 32 < kEnd) {
            __builtin_prefetch(&vp[(size_t)(kt + 32) * DKH + (tid & 63)], 0, 0);
        }
#ifdef USE_TDM
        __builtin_amdgcn_s_wait_tensorcnt(0);   // wave 0 waits; others no-op
#endif
        __syncthreads();

        if (kt <= qbase + 15) {          // tile intersects this wave's causal range
            // Scores: 16 q-rows x 32 keys = two 16x16 D fragments.
            FragC sc[2];
            #pragma unroll
            for (int ns = 0; ns < 2; ++ns) {
                #pragma unroll
                for (int i = 0; i < 8; ++i) sc[ns].f[i] = 0.0f;
                #pragma unroll
                for (int t = 0; t < 2; ++t) {
                    FragA kb;
                    int key = ns * 16 + m16;
                    kb.q[0] = *reinterpret_cast<const uint4*>(&Ksh[key][t * 32 + half * 8]);
                    kb.q[1] = *reinterpret_cast<const uint4*>(&Ksh[key][t * 32 + 16 + half * 8]);
                    sc[ns].v = __builtin_amdgcn_wmma_f32_16x16x32_bf16(
                        false, qa[t].v, false, kb.v, (short)0, sc[ns].v, false, false);
                }
            }

            // Scale (1/sqrt(64) = 0.125) + causal mask.
            float sm[2][8];
            #pragma unroll
            for (int ns = 0; ns < 2; ++ns) {
                int key = kt + ns * 16 + m16;
                #pragma unroll
                for (int r = 0; r < 8; ++r) {
                    int qrow = qbase + r + half * 8;
                    sm[ns][r] = (key <= qrow) ? sc[ns].f[r] * 0.125f : -1e30f;
                }
            }

            // Online softmax (row stats live uniformly per 16-lane half).
            float alpha[8];
            #pragma unroll
            for (int r = 0; r < 8; ++r) {
                float rm = fmaxf(sm[0][r], sm[1][r]);
                #pragma unroll
                for (int off = 1; off < 16; off <<= 1)
                    rm = fmaxf(rm, __shfl_xor(rm, off, 32));
                float mn = fmaxf(mrow[r], rm);
                alpha[r] = __expf(mrow[r] - mn);
                mrow[r]  = mn;
                float p0 = __expf(sm[0][r] - mn);
                float p1 = __expf(sm[1][r] - mn);
                sm[0][r] = p0;
                sm[1][r] = p1;
                float rs = p0 + p1;
                #pragma unroll
                for (int off = 1; off < 16; off <<= 1)
                    rs += __shfl_xor(rs, off, 32);
                lrow[r] = lrow[r] * alpha[r] + rs;
            }
            #pragma unroll
            for (int g = 0; g < 4; ++g)
                #pragma unroll
                for (int r = 0; r < 8; ++r) acc[g].f[r] *= alpha[r];

            // Transpose P (D-layout -> A-layout) through per-wave LDS.
            #pragma unroll
            for (int ns = 0; ns < 2; ++ns)
                #pragma unroll
                for (int r = 0; r < 8; ++r)
                    Psh[wave][r + half * 8][ns * 16 + m16] = (__bf16)sm[ns][r];
            asm volatile("s_wait_dscnt 0" ::: "memory");

            FragA pa;
            pa.q[0] = *reinterpret_cast<const uint4*>(&Psh[wave][m16][half * 8]);
            pa.q[1] = *reinterpret_cast<const uint4*>(&Psh[wave][m16][16 + half * 8]);

            // acc += P @ V  (V^T in LDS makes B fragments contiguous runs).
            #pragma unroll
            for (int g = 0; g < 4; ++g) {
                FragA vb;
                int dk = g * 16 + m16;
                vb.q[0] = *reinterpret_cast<const uint4*>(&VTsh[dk][half * 8]);
                vb.q[1] = *reinterpret_cast<const uint4*>(&VTsh[dk][16 + half * 8]);
                acc[g].v = __builtin_amdgcn_wmma_f32_16x16x32_bf16(
                    false, pa.v, false, vb.v, (short)0, acc[g].v, false, false);
            }
        }
    }

    // Epilogue: divide by row sums, write fp32 [B][S][D].
    #pragma unroll
    for (int g = 0; g < 4; ++g) {
        int col = h * DKH + g * 16 + m16;
        #pragma unroll
        for (int r = 0; r < 8; ++r) {
            int row = qbase + r + half * 8;
            out[((size_t)b * SLEN + row) * DMODEL + col] = acc[g].f[r] / lrow[r];
        }
    }
}

extern "C" void kernel_launch(void* const* d_in, const int* in_sizes, int n_in,
                              void* d_out, int out_size, void* d_ws, size_t ws_size,
                              hipStream_t stream) {
    const float* x  = (const float*)d_in[0];
    const float* Wq = (const float*)d_in[1];
    const float* bq = (const float*)d_in[2];
    const float* Wk = (const float*)d_in[3];
    const float* bk = (const float*)d_in[4];
    const float* Wv = (const float*)d_in[5];
    const float* bv = (const float*)d_in[6];
    __bf16* qkv = (__bf16*)d_ws;     // 3 * B*H*S*DK bf16 = ~25 MB

    dim3 g1(BSZ * SLEN / 64, DMODEL / 64, 3);   // 64 x 16 x 3
    qkv_proj_kernel<<<g1, dim3(128), 0, stream>>>(x, Wq, bq, Wk, bk, Wv, bv, qkv);

    dim3 g2(SLEN / 64, BSZ * NHEAD);            // 32 x 32
    flash_attn_kernel<<<g2, dim3(128), 0, stream>>>(qkv, (float*)d_out);
}
